// Cov_EBFLayer_19610820673639
// MI455X (gfx1250) — matrix operand order repaired and checked by reading, not compiled, not run on hardware
//
#include <hip/hip_runtime.h>
#include <hip/hip_bf16.h>

// out[b,o] = exp(-||beta[o]^T (center[o]-x[b])||^2), B=8192, D=64, O=256
// v2: prep kernel pre-splits betas to f16 hi/lo (padded, e-major) + cproj in d_ws;
//     main kernel double-buffers per-o tiles into LDS with async-to-LDS DMA and
//     runs the f16-split WMMA GEMM + (cp - xp)^2 reduce + exp.

typedef __attribute__((ext_vector_type(16))) _Float16 v16h;
typedef __attribute__((ext_vector_type(8)))  _Float16 v8h;
typedef __attribute__((ext_vector_type(8)))  float    v8f;
typedef __attribute__((ext_vector_type(4)))  float    v4f;

#define Bsz 8192
#define Dd  64
#define Osz 256
#define BSTRIDE 72                 // padded row stride (f16 elems), 144B rows
#define HL_HALFS (Dd * BSTRIDE)    // 4608 halves = 9216 B per hi (or lo) tile
#define HL_BYTES (2 * HL_HALFS * 2)// 18432 B: hi + lo
#define PER_O (HL_BYTES + Dd * 4)  // + 256 B cproj = 18688 B per o (16B multiple)
#define CHUNKS (PER_O / 16)        // 1168 x b128 per o
#define NO 8                       // o's per block in main kernel

#ifndef ASYNC_STAGE
#define ASYNC_STAGE 1
#endif

__device__ __forceinline__ void stage_tile(uint8_t* dst_lds,
                                           const uint8_t* __restrict__ src,
                                           int tid) {
#if ASYNC_STAGE
    uint32_t ldsbase = (uint32_t)(size_t)dst_lds;   // low 32 bits = LDS offset
    for (int i = tid; i < CHUNKS; i += 256) {
        asm volatile("global_load_async_to_lds_b128 %0, %1, %2"
                     :: "v"(ldsbase + (uint32_t)(i * 16)), "v"(i * 16), "s"(src)
                     : "memory");
    }
#else
    for (int i = tid; i < CHUNKS; i += 256)
        ((int4*)dst_lds)[i] = ((const int4*)src)[i];
#endif
}

__device__ __forceinline__ void stage_wait() {
#if ASYNC_STAGE
#if defined(__has_builtin) && __has_builtin(__builtin_amdgcn_s_wait_asynccnt)
    __builtin_amdgcn_s_wait_asynccnt(0);
#else
    asm volatile("s_wait_asynccnt 0x0" ::: "memory");
#endif
#endif
}

// ---------------- prep: betas -> transposed f16 hi/lo (+pad) and cproj ----------------
__global__ __launch_bounds__(256) void ebf_prep(
    const float* __restrict__ centers,  // [O, 64]
    const float* __restrict__ betas,    // [O, 64, 64]
    uint8_t* __restrict__ ws)
{
    __shared__ float sF32[Dd * Dd];
    const int tid = threadIdx.x, o = blockIdx.x;

    const float* bet = betas + (size_t)o * (Dd * Dd);
    for (int i = tid; i < Dd * Dd; i += 256) sF32[i] = bet[i];
    __syncthreads();

    _Float16* dstH = (_Float16*)(ws + (size_t)o * PER_O);
    _Float16* dstL = dstH + HL_HALFS;
    float*    dstC = (float*)(ws + (size_t)o * PER_O + HL_BYTES);

    for (int j = tid; j < HL_HALFS; j += 256) {   // coalesced f16 stores, e-major
        int e = j / BSTRIDE, d = j - e * BSTRIDE;
        if (d < Dd) {
            float v = sF32[d * Dd + e];
            _Float16 h = (_Float16)v;
            dstH[j] = h;
            dstL[j] = (_Float16)(v - (float)h);
        }
    }
    if (tid < Dd) {                               // cproj[e] = sum_d cen[d]*beta[d,e]
        float acc = 0.f;
        const float* cen = centers + o * Dd;
        #pragma unroll 8
        for (int d = 0; d < Dd; ++d) acc += cen[d] * sF32[d * Dd + tid];
        dstC[tid] = acc;
    }
}

// ---------------- main: x-split once, stream 8 o's with double-buffered DMA ----------------
__global__ __launch_bounds__(256) void ebf_main(
    const float* __restrict__ x,    // [B, 64]
    const uint8_t* __restrict__ ws, // per-o packed tiles
    float* __restrict__ out)        // [B, O]
{
    __shared__ __align__(16) uint8_t sMem[2][PER_O];
    __shared__ float sOut[128][NO];

    const int tid  = threadIdx.x;
    const int lane = tid & 31;
    const int wave = tid >> 5;
    const int o0   = blockIdx.y * NO;

    // kick off DMA of first beta tile, then build A while it flies
    stage_tile(sMem[0], ws + (size_t)o0 * PER_O, tid);

    // ---- A = x tile (16 rows x 64 K) as f16 hi/lo in WMMA A-layout ----
    // lane<16: M=lane, K chunks {0-7,16-23}; lane>=16: M=lane-16, K {8-15,24-31};
    // second v16h covers K 32..63.
    const int row = blockIdx.x * 128 + wave * 16 + (lane & 15);
    const v4f* xrow = (const v4f*)(x + (size_t)row * Dd);
    const int kb = (lane & 16) ? 8 : 0;

    _Float16 hi[32], lo[32];
    #pragma unroll
    for (int ci = 0; ci < 4; ++ci) {
        const int c = kb + ci * 16;
        v4f x0 = xrow[c >> 2];
        v4f x1 = xrow[(c >> 2) + 1];
        #pragma unroll
        for (int j = 0; j < 4; ++j) {
            float d0 = x0[j], d1 = x1[j];
            _Float16 h0 = (_Float16)d0;
            _Float16 h1 = (_Float16)d1;
            hi[ci * 8 + j]     = h0;  lo[ci * 8 + j]     = (_Float16)(d0 - (float)h0);
            hi[ci * 8 + 4 + j] = h1;  lo[ci * 8 + 4 + j] = (_Float16)(d1 - (float)h1);
        }
    }
    v16h Ah[2], Al[2];
    #pragma unroll
    for (int kh = 0; kh < 2; ++kh) {
        #pragma unroll
        for (int j = 0; j < 8; ++j) {
            Ah[kh][j]     = hi[kh * 16 + j];
            Ah[kh][8 + j] = hi[kh * 16 + 8 + j];
            Al[kh][j]     = lo[kh * 16 + j];
            Al[kh][8 + j] = lo[kh * 16 + 8 + j];
        }
    }

    stage_wait();
    __syncthreads();

    const int bcol = lane & 15;
    const int krow = (lane & 16) ? 8 : 0;

    #pragma clang loop unroll(disable)
    for (int k = 0; k < NO; ++k) {
        uint8_t* cur = sMem[k & 1];
        if (k + 1 < NO)  // prefetch next o during this o's WMMAs
            stage_tile(sMem[(k + 1) & 1], ws + (size_t)(o0 + k + 1) * PER_O, tid);

        const _Float16* sBh = (const _Float16*)cur;
        const _Float16* sBl = (const _Float16*)(cur + HL_HALFS * 2);
        const float*    sCp = (const float*)(cur + HL_BYTES);

        float cpv[4];
        #pragma unroll
        for (int t = 0; t < 4; ++t) cpv[t] = sCp[t * 16 + bcol];

        v8f C[4];
        #pragma unroll
        for (int t = 0; t < 4; ++t) {
            v8f z = {0.f, 0.f, 0.f, 0.f, 0.f, 0.f, 0.f, 0.f};
            C[t] = z;
        }

        #pragma unroll
        for (int kh = 0; kh < 2; ++kh) {
            const int koff = krow + kh * 32;
            #pragma unroll
            for (int pr = 0; pr < 3; ++pr) {   // (Ah,Bh), (Ah,Bl), (Al,Bh)
                v16h a = (pr == 2) ? Al[kh] : Ah[kh];
                const _Float16* bb = (pr == 1) ? sBl : sBh;
                #pragma unroll
                for (int t = 0; t < 4; ++t) {
                    const v8h* bp = (const v8h*)(bb + (t * 16 + bcol) * BSTRIDE + koff);
                    v8h b0 = bp[0], b1 = bp[2];   // 2x ds_load_b128
                    v16h bm;
                    #pragma unroll
                    for (int j = 0; j < 8; ++j) { bm[j] = b0[j]; bm[8 + j] = b1[j]; }
                    C[t] = __builtin_amdgcn_wmma_f32_16x16x32_f16(
                        false, a, false, bm, (short)0, C[t], false, false);
                }
            }
        }

        // quad[m] = sum_e (cp[e] - xp[m,e])^2
        float q[8] = {0.f, 0.f, 0.f, 0.f, 0.f, 0.f, 0.f, 0.f};
        #pragma unroll
        for (int t = 0; t < 4; ++t) {
            #pragma unroll
            for (int r = 0; r < 8; ++r) {
                float y = cpv[t] - C[t][r];
                q[r] += y * y;
            }
        }
        #pragma unroll
        for (int r = 0; r < 8; ++r) {
            q[r] += __shfl_xor(q[r], 1, 32);
            q[r] += __shfl_xor(q[r], 2, 32);
            q[r] += __shfl_xor(q[r], 4, 32);
            q[r] += __shfl_xor(q[r], 8, 32);
        }

        const int lm = lane & 15;
        if (lm < 8) {
            float res = q[0];
            #pragma unroll
            for (int r = 1; r < 8; ++r)
                if (lm == r) res = q[r];
            const int m = (lane < 16) ? lm : (lm + 8);
            sOut[wave * 16 + m][k] = __expf(-res);
        }

        stage_wait();
        __syncthreads();
    }

    // coalesced output: each of 128 threads writes its row's 8 o's (32B)
    if (tid < 128) {
        const int orow = blockIdx.x * 128 + tid;
        float4* dst = (float4*)(out + (size_t)orow * Osz + o0);
        const float4* srcv = (const float4*)sOut[tid];
        dst[0] = srcv[0];
        dst[1] = srcv[1];
    }
}

// ---------------- fallback (no workspace): fused single kernel, 1 o per block ----------------
__global__ __launch_bounds__(256) void ebf_fallback(
    const float* __restrict__ x,
    const float* __restrict__ centers,
    const float* __restrict__ betas,
    float* __restrict__ out)
{
    __shared__ _Float16 sBh[Dd * BSTRIDE];
    __shared__ _Float16 sBl[Dd * BSTRIDE];
    __shared__ float    sCen[Dd];

    const int tid  = threadIdx.x;
    const int o    = blockIdx.y;
    const int lane = tid & 31;
    const int wave = tid >> 5;

    const float* bet = betas + (size_t)o * (Dd * Dd);
    for (int i = tid; i < Dd * Dd; i += 256) {
        float v = bet[i];
        int d = i >> 6, e = i & 63;
        _Float16 h = (_Float16)v;
        sBh[e * BSTRIDE + d] = h;
        sBl[e * BSTRIDE + d] = (_Float16)(v - (float)h);
    }
    if (tid < Dd) sCen[tid] = centers[o * Dd + tid];
    __syncthreads();

    const int row = blockIdx.x * 128 + wave * 16 + (lane & 15);
    const v4f* xrow = (const v4f*)(x + (size_t)row * Dd);
    const int kb = (lane & 16) ? 8 : 0;

    _Float16 hi[32], lo[32];
    #pragma unroll
    for (int ci = 0; ci < 4; ++ci) {
        const int c = kb + ci * 16;
        v4f x0 = xrow[c >> 2];
        v4f x1 = xrow[(c >> 2) + 1];
        #pragma unroll
        for (int j = 0; j < 4; ++j) {
            float d0 = sCen[c + j]     - x0[j];
            float d1 = sCen[c + 4 + j] - x1[j];
            _Float16 h0 = (_Float16)d0;
            _Float16 h1 = (_Float16)d1;
            hi[ci * 8 + j]     = h0;  lo[ci * 8 + j]     = (_Float16)(d0 - (float)h0);
            hi[ci * 8 + 4 + j] = h1;  lo[ci * 8 + 4 + j] = (_Float16)(d1 - (float)h1);
        }
    }
    v16h Ah[2], Al[2];
    #pragma unroll
    for (int kh = 0; kh < 2; ++kh) {
        #pragma unroll
        for (int j = 0; j < 8; ++j) {
            Ah[kh][j]     = hi[kh * 16 + j];
            Ah[kh][8 + j] = hi[kh * 16 + 8 + j];
            Al[kh][j]     = lo[kh * 16 + j];
            Al[kh][8 + j] = lo[kh * 16 + 8 + j];
        }
    }

    v8f C[4];
    #pragma unroll
    for (int t = 0; t < 4; ++t) {
        v8f z = {0.f, 0.f, 0.f, 0.f, 0.f, 0.f, 0.f, 0.f};
        C[t] = z;
    }
    const int bcol = lane & 15;
    const int krow = (lane & 16) ? 8 : 0;

    #pragma unroll
    for (int kh = 0; kh < 2; ++kh) {
        const int koff = krow + kh * 32;
        #pragma unroll
        for (int pr = 0; pr < 3; ++pr) {
            v16h a = (pr == 2) ? Al[kh] : Ah[kh];
            const _Float16* bb = (pr == 1) ? sBl : sBh;
            #pragma unroll
            for (int t = 0; t < 4; ++t) {
                const v8h* bp = (const v8h*)(bb + (t * 16 + bcol) * BSTRIDE + koff);
                v8h b0 = bp[0], b1 = bp[2];
                v16h bm;
                #pragma unroll
                for (int j = 0; j < 8; ++j) { bm[j] = b0[j]; bm[8 + j] = b1[j]; }
                C[t] = __builtin_amdgcn_wmma_f32_16x16x32_f16(
                    false, a, false, bm, (short)0, C[t], false, false);
            }
        }
    }

    float q[8] = {0.f, 0.f, 0.f, 0.f, 0.f, 0.f, 0.f, 0.f};
    #pragma unroll
    for (int t = 0; t < 4; ++t) {
        #pragma unroll
        for (int r = 0; r < 8; ++r) q[r] += C[t][r] * C[t][r];
    }
    #pragma unroll
    for (int r = 0; r < 8; ++r) {
        q[r] += __shfl_xor(q[r], 1, 32);
        q[r] += __shfl_xor(q[r], 2, 32);
        q[r] += __shfl_xor(q[r], 4, 32);
        q[r] += __shfl_xor(q[r], 8, 32);
    }
    const int lm = lane & 15;
    if (lm < 8) {
        float res = q[0];
        #pragma unroll
        for (int r = 1; r < 8; ++r)
            if (lm == r) res = q[r];
        const int m = (lane < 16) ? lm : (lm + 8);
        const int orow = blockIdx.x * 128 + wave * 16 + m;
        out[(size_t)orow * Osz + o] = __expf(-res);
    }
}

extern "C" void kernel_launch(void* const* d_in, const int* in_sizes, int n_in,
                              void* d_out, int out_size, void* d_ws, size_t ws_size,
                              hipStream_t stream) {
    const float* x       = (const float*)d_in[0];  // [8192, 64]
    const float* centers = (const float*)d_in[1];  // [256, 1, 64]
    const float* betas   = (const float*)d_in[2];  // [256, 64, 64]
    float* out = (float*)d_out;                    // [8192, 256]

    const size_t ws_need = (size_t)Osz * PER_O;    // ~4.6 MB
    if (d_ws != nullptr && ws_size >= ws_need) {
        ebf_prep<<<dim3(Osz), 256, 0, stream>>>(centers, betas, (uint8_t*)d_ws);
        ebf_main<<<dim3(Bsz / 128, Osz / NO), 256, 0, stream>>>(
            x, (const uint8_t*)d_ws, out);
    } else {
        ebf_fallback<<<dim3(Bsz / 128, Osz), 256, 0, stream>>>(x, centers, betas, out);
    }
}
